// PosteriorModule_62010737819869
// MI455X (gfx1250) — compile-verified, matrix operand order checked by reading, not run on hardware
//
#include <hip/hip_runtime.h>
#include <hip/hip_bf16.h>

// ---------------------------------------------------------------------------
// PosteriorModule forward for MI455X (gfx1250, wave32, WMMA)
// B=16 T=1024 D=256 H=8 DH=32 L=64 F=32   SCALE = 16
//
// Restructured vs reference:
//   * only first F rows of post_mha output are needed -> Q proj on 512 rows
//   * K/V projections of h_C hoisted out of the scan (loop invariant)
//   * K/V of w_buf maintained incrementally (one new row per step)
// GEMMs: v_wmma_f32_16x16x32_bf16 (f32 accumulate).
// Big projections: 4 waves/block share an async-staged LDS weight panel.
// ---------------------------------------------------------------------------

#define BB 16
#define TT 1024
#define DD 256
#define HH 8
#define DH 32
#define LL 64
#define FF 32
#define INV_SCALE (1.0f/16.0f)

typedef __attribute__((ext_vector_type(16))) __bf16 v16bf;
typedef __attribute__((ext_vector_type(8)))  float  v8f;
typedef __attribute__((ext_vector_type(4)))  int    v4i;

__device__ __forceinline__ v8f wmma_bf16(v16bf a, v16bf b, v8f c) {
    return __builtin_amdgcn_wmma_f32_16x16x32_bf16(
               false, a, false, b, (short)0, c, false, false);
}

// ---- CDNA5 async global->LDS copy (ASYNCcnt), with safe fallback ----------
#if __has_builtin(__builtin_amdgcn_global_load_async_to_lds_b128)
#define HAVE_ASYNC_COPY 1
#else
#define HAVE_ASYNC_COPY 0
#endif

__device__ __forceinline__ void async_copy_b128(const float* g, float* l) {
#if HAVE_ASYNC_COPY
    typedef __attribute__((address_space(1))) v4i* g4p;   // global int4*
    typedef __attribute__((address_space(3))) v4i* l4p;   // LDS int4*
    __builtin_amdgcn_global_load_async_to_lds_b128(
        (g4p)(v4i*)const_cast<float*>(g), (l4p)(v4i*)l, 0, 0);
#else
    *(float4*)l = *(const float4*)g;
#endif
}

__device__ __forceinline__ void async_wait_all() {
#if HAVE_ASYNC_COPY
#if __has_builtin(__builtin_amdgcn_s_wait_asynccnt)
    __builtin_amdgcn_s_wait_asynccnt(0);
#else
    asm volatile("s_wait_asynccnt 0" ::: "memory");
#endif
#endif
}

// 16x32 bf16 A-fragment, K contiguous in memory (row-major along K).
// Lane layout (ISA 7.12.2): lanes 0-15 hold K {0..7,16..23}, lanes 16-31
// hold K {8..15,24..31}  ->  two contiguous 8-float runs per lane.
__device__ __forceinline__ v16bf load_frag_rowK(const float* __restrict__ p, int half) {
    const float4 a0 = *(const float4*)(p + 8 * half);
    const float4 a1 = *(const float4*)(p + 8 * half + 4);
    const float4 b0 = *(const float4*)(p + 16 + 8 * half);
    const float4 b1 = *(const float4*)(p + 16 + 8 * half + 4);
    v16bf f;
    f[0]=(__bf16)a0.x;  f[1]=(__bf16)a0.y;  f[2]=(__bf16)a0.z;  f[3]=(__bf16)a0.w;
    f[4]=(__bf16)a1.x;  f[5]=(__bf16)a1.y;  f[6]=(__bf16)a1.z;  f[7]=(__bf16)a1.w;
    f[8]=(__bf16)b0.x;  f[9]=(__bf16)b0.y;  f[10]=(__bf16)b0.z; f[11]=(__bf16)b0.w;
    f[12]=(__bf16)b1.x; f[13]=(__bf16)b1.y; f[14]=(__bf16)b1.z; f[15]=(__bf16)b1.w;
    return f;
}

// 32x16 bf16 B-fragment from K x N row-major B; LDB compile-time so every
// kk*LDB folds into the global_load immediate offset.  p = B + k0*LDB + col.
template<int LDB>
__device__ __forceinline__ v16bf load_frag_colK(const float* __restrict__ p, int half) {
    v16bf f;
    #pragma unroll
    for (int j = 0; j < 16; ++j) {
        const int kk = (j < 8) ? (8 * half + j) : (8 + 8 * half + j);
        f[j] = (__bf16)p[kk * LDB];
    }
    return f;
}

// ---------------------------------------------------------------------------
// Big-M GEMM (M % 64 == 0, N = 256 weights, K = 256): 128 threads = 4 waves,
// 64x64 output tile per block.  The 32x64 B panel for each k-step is staged
// in LDS once per block via async global->LDS copies and shared by all waves.
// grid = (N/64, M/64)
// ---------------------------------------------------------------------------
template<int ROWMAP>
__global__ __launch_bounds__(128) void wmma_gemm_big(
    const float* __restrict__ A, const float* __restrict__ Bm,
    const float* __restrict__ bias, float* __restrict__ C,
    int lda, int ldc)
{
    __shared__ float bpan[32 * 64];                 // 8 KB B panel
    const int tid  = threadIdx.x;
    const int lane = tid & 31, wave = tid >> 5;
    const int half = lane >> 4, rc = lane & 15;
    const int tN   = blockIdx.x * 64;
    const int tM   = blockIdx.y * 64 + wave * 16;

    int arow = tM + rc;
    if (ROWMAP) arow = (arow / FF) * TT + (arow % FF);
    const float* __restrict__ Arow = A + (long)arow * lda;

    v8f acc[4] = {{}, {}, {}, {}};
    for (int k0 = 0; k0 < DD; k0 += 32) {
        // stage B[k0..k0+32) x [tN..tN+64) -> LDS:  512 x 16B chunks
        #pragma unroll
        for (int it = 0; it < 4; ++it) {
            const int idx   = it * 128 + tid;
            const int row   = idx >> 4;             // 0..31
            const int chunk = idx & 15;             // 16B chunks in a 64-col row
            async_copy_b128(Bm + (long)(k0 + row) * DD + tN + chunk * 4,
                            &bpan[row * 64 + chunk * 4]);
        }
        async_wait_all();
        __syncthreads();

        if (k0 + 32 < DD) __builtin_prefetch(Arow + k0 + 32, 0, 1);
        const v16bf af = load_frag_rowK(Arow + k0, half);
        #pragma unroll
        for (int t = 0; t < 4; ++t) {
            v16bf bf;
            #pragma unroll
            for (int j = 0; j < 16; ++j) {
                const int kk = (j < 8) ? (8 * half + j) : (8 + 8 * half + j);
                bf[j] = (__bf16)bpan[kk * 64 + t * 16 + rc];
            }
            acc[t] = wmma_bf16(af, bf, acc[t]);
        }
        __syncthreads();
    }
    #pragma unroll
    for (int t = 0; t < 4; ++t) {
        const int n = tN + t * 16 + rc;
        const float bv = bias ? bias[n] : 0.0f;
        #pragma unroll
        for (int r = 0; r < 8; ++r)
            C[(long)(tM + r + 8 * half) * ldc + n] = acc[t][r] + bv;
    }
}

// ---------------------------------------------------------------------------
// Small GEMM: C[M,N] = A @ B + bias, one wave per 16x64 tile (M = 16 uses).
// grid = (N/64, M/16).  K, LDB compile-time.
// ---------------------------------------------------------------------------
template<int K, int LDB>
__global__ __launch_bounds__(32) void wmma_gemm(
    const float* __restrict__ A, const float* __restrict__ Bm,
    const float* __restrict__ bias, float* __restrict__ C,
    int lda, int ldc)
{
    const int lane = threadIdx.x & 31;
    const int half = lane >> 4;
    const int rc   = lane & 15;
    const int tN   = blockIdx.x * 64;
    const int tM   = blockIdx.y * 16;

    const float* __restrict__ Arow = A + (long)(tM + rc) * lda;
    const float* __restrict__ Bcol = Bm + tN + rc;

    v8f acc[4] = {{}, {}, {}, {}};
    #pragma unroll
    for (int k0 = 0; k0 < K; k0 += 32) {
        const v16bf af = load_frag_rowK(Arow + k0, half);
        const float* __restrict__ Bk = Bcol + (long)k0 * LDB;
        const v16bf b0 = load_frag_colK<LDB>(Bk,      half);
        const v16bf b1 = load_frag_colK<LDB>(Bk + 16, half);
        const v16bf b2 = load_frag_colK<LDB>(Bk + 32, half);
        const v16bf b3 = load_frag_colK<LDB>(Bk + 48, half);
        acc[0] = wmma_bf16(af, b0, acc[0]);
        acc[1] = wmma_bf16(af, b1, acc[1]);
        acc[2] = wmma_bf16(af, b2, acc[2]);
        acc[3] = wmma_bf16(af, b3, acc[3]);
    }
    #pragma unroll
    for (int t = 0; t < 4; ++t) {
        const int n = tN + t * 16 + rc;
        const float bv = bias ? bias[n] : 0.0f;
        #pragma unroll
        for (int r = 0; r < 8; ++r)
            C[(long)(tM + r + 8 * half) * ldc + n] = acc[t][r] + bv;
    }
}

// ---------------------------------------------------------------------------
// Post-MHA scores: S[b,h] (F x T) = Q32[b,:,h] @ KT[b,:,h]^T / SCALE
// grid (T/64, F/16, B*H); both operands row-major along K -> vector loads.
// ---------------------------------------------------------------------------
__global__ __launch_bounds__(32) void attn_scores(
    const float* __restrict__ Q32, const float* __restrict__ KT,
    float* __restrict__ S)
{
    const int z = blockIdx.z, b = z / HH, h = z % HH;
    const int lane = threadIdx.x & 31, half = lane >> 4, rc = lane & 15;
    const int tN = blockIdx.x * 64, tM = blockIdx.y * 16;

    const float* Ap = Q32 + (long)b * FF * DD + h * DH + (long)(tM + rc) * DD;
    const float* Bp = KT  + (long)b * TT * DD + h * DH;

    const v16bf af = load_frag_rowK(Ap, half);           // K = DH = 32, one step
    v8f acc[4] = {{}, {}, {}, {}};
    #pragma unroll
    for (int t = 0; t < 4; ++t) {
        const v16bf bf = load_frag_rowK(Bp + (long)(tN + t * 16 + rc) * DD, half);
        acc[t] = wmma_bf16(af, bf, acc[t]);
    }
    float* Sp = S + (long)z * FF * TT;
    #pragma unroll
    for (int t = 0; t < 4; ++t)
        #pragma unroll
        for (int r = 0; r < 8; ++r)
            Sp[(long)(tM + r + 8 * half) * TT + tN + t * 16 + rc] = acc[t][r] * INV_SCALE;
}

// O32[b,:,h] (F x DH) = P[b,h] (F x T) @ VT[b,:,h];  grid (F/16, B*H)
__global__ __launch_bounds__(32) void attn_av(
    const float* __restrict__ S, const float* __restrict__ VT,
    float* __restrict__ O32)
{
    const int z = blockIdx.y, b = z / HH, h = z % HH;
    const int lane = threadIdx.x & 31, half = lane >> 4, rc = lane & 15;
    const int tM = blockIdx.x * 16;

    const float* Ap = S  + (long)z * FF * TT + (long)(tM + rc) * TT;
    const float* Bp = VT + (long)b * TT * DD + h * DH + rc;

    v8f acc0 = {}, acc1 = {};
    #pragma unroll 4
    for (int k0 = 0; k0 < TT; k0 += 32) {
        const v16bf af = load_frag_rowK(Ap + k0, half);
        const v16bf b0 = load_frag_colK<DD>(Bp + (long)k0 * DD,      half);
        const v16bf b1 = load_frag_colK<DD>(Bp + (long)k0 * DD + 16, half);
        acc0 = wmma_bf16(af, b0, acc0);
        acc1 = wmma_bf16(af, b1, acc1);
    }
    float* Op = O32 + (long)b * FF * DD + h * DH;
    #pragma unroll
    for (int r = 0; r < 8; ++r) {
        Op[(long)(tM + r + 8 * half) * DD + rc]      = acc0[r];
        Op[(long)(tM + r + 8 * half) * DD + 16 + rc] = acc1[r];
    }
}

// one wave per row
__global__ __launch_bounds__(32) void softmax_rows(float* __restrict__ S, int ncols)
{
    float* p = S + (long)blockIdx.x * ncols;
    const int lane = threadIdx.x & 31;
    float m = -1e30f;
    for (int c = lane; c < ncols; c += 32) m = fmaxf(m, p[c]);
    for (int o = 16; o; o >>= 1) m = fmaxf(m, __shfl_xor(m, o, 32));
    float s = 0.f;
    for (int c = lane; c < ncols; c += 32) { float e = __expf(p[c] - m); p[c] = e; s += e; }
    for (int o = 16; o; o >>= 1) s += __shfl_xor(s, o, 32);
    const float inv = 1.0f / s;
    for (int c = lane; c < ncols; c += 32) p[c] *= inv;
}

// y = LN(x + r) * g + b ; one wave per batch row, D = 256
__global__ __launch_bounds__(32) void ln_resid(
    const float* __restrict__ x, int ldx, const float* __restrict__ r,
    const float* __restrict__ g, const float* __restrict__ bta,
    float* __restrict__ out, int ldout)
{
    const int b = blockIdx.x, lane = threadIdx.x & 31;
    const float* xp = x + (long)b * ldx;
    const float* rp = r + (long)b * DD;
    float v[8]; float mean = 0.f;
    #pragma unroll
    for (int j = 0; j < 8; ++j) { v[j] = xp[lane + 32 * j] + rp[lane + 32 * j]; mean += v[j]; }
    for (int o = 16; o; o >>= 1) mean += __shfl_xor(mean, o, 32);
    mean *= (1.0f / DD);
    float var = 0.f;
    #pragma unroll
    for (int j = 0; j < 8; ++j) { float d = v[j] - mean; var += d * d; }
    for (int o = 16; o; o >>= 1) var += __shfl_xor(var, o, 32);
    var *= (1.0f / DD);
    const float inv = rsqrtf(var + 1e-5f);
    float* op = out + (long)b * ldout;
    #pragma unroll
    for (int j = 0; j < 8; ++j) {
        const int c = lane + 32 * j;
        op[c] = (v[j] - mean) * inv * g[c] + bta[c];
    }
}

// prior_mha1 attention: 1 query, nk = i+1 keys from incremental Kbuf/Vbuf
__global__ __launch_bounds__(32) void attn_small(
    const float* __restrict__ q, const float* __restrict__ Kb,
    const float* __restrict__ Vb, float* __restrict__ o, int nk)
{
    const int z = blockIdx.x, b = z / HH, h = z % HH, lane = threadIdx.x & 31;
    __shared__ float lg[FF + 1];
    __shared__ float qv[DH];
    qv[lane] = q[b * DD + h * DH + lane];
    __syncthreads();
    const float* Kp = Kb + (long)b * (FF + 1) * DD + h * DH;
    const float* Vp = Vb + (long)b * (FF + 1) * DD + h * DH;
    for (int k = lane; k < nk; k += 32) {
        float acc = 0.f;
        #pragma unroll 8
        for (int d = 0; d < DH; ++d) acc += qv[d] * Kp[(long)k * DD + d];
        lg[k] = acc * INV_SCALE;
    }
    __syncthreads();
    float m = -1e30f;
    for (int k = 0; k < nk; ++k) m = fmaxf(m, lg[k]);
    float s = 0.f;
    for (int k = 0; k < nk; ++k) s += __expf(lg[k] - m);
    const float inv = 1.0f / s;
    float acc = 0.f;
    for (int k = 0; k < nk; ++k) acc += __expf(lg[k] - m) * Vp[(long)k * DD + lane];
    o[b * DD + h * DH + lane] = acc * inv;
}

// prior_mha2 attention: 1 query vs 1024 hoisted h_C keys; one wave per (b,h)
__global__ __launch_bounds__(32) void attn_long(
    const float* __restrict__ q, const float* __restrict__ Kc,
    const float* __restrict__ Vc, float* __restrict__ o)
{
    const int z = blockIdx.x, b = z / HH, h = z % HH, lane = threadIdx.x & 31;
    __shared__ float lg[TT];
    __shared__ float qv[DH];
    qv[lane] = q[b * DD + h * DH + lane];
    __syncthreads();
    const float* Kp = Kc + (long)b * TT * DD + h * DH;
    const float* Vp = Vc + (long)b * TT * DD + h * DH;
    float lmax = -1e30f;
    for (int k0 = 0; k0 < TT; k0 += 32) {
        const int k = k0 + lane;
        float acc = 0.f;
        #pragma unroll 8
        for (int d = 0; d < DH; ++d) acc += qv[d] * Kp[(long)k * DD + d];
        acc *= INV_SCALE;
        lg[k] = acc;
        lmax = fmaxf(lmax, acc);
    }
    for (int ofs = 16; ofs; ofs >>= 1) lmax = fmaxf(lmax, __shfl_xor(lmax, ofs, 32));
    __syncthreads();
    float lsum = 0.f;
    for (int k = lane; k < TT; k += 32) { float e = __expf(lg[k] - lmax); lg[k] = e; lsum += e; }
    for (int ofs = 16; ofs; ofs >>= 1) lsum += __shfl_xor(lsum, ofs, 32);
    __syncthreads();
    const float inv = 1.0f / lsum;
    float acc = 0.f;
    for (int k = 0; k < TT; ++k) acc += lg[k] * Vp[(long)k * DD + lane];
    o[b * DD + h * DH + lane] = acc * inv;
}

// -------------------------- small elementwise kernels ----------------------
__global__ void bcast_w0(const float* __restrict__ w0, float* __restrict__ wbuf)
{
    const int idx = blockIdx.x * blockDim.x + threadIdx.x;
    if (idx >= BB * DD) return;
    const int b = idx / DD, d = idx % DD;
    wbuf[(long)b * (FF + 1) * DD + d] = w0[d];
}

__global__ void copy_ki(const float* __restrict__ kall32, float* __restrict__ whk, int i)
{
    const int idx = blockIdx.x * blockDim.x + threadIdx.x;
    if (idx >= BB * DD) return;
    const int b = idx / DD, d = idx % DD;
    whk[b * 2 * DD + DD + d] = kall32[((long)b * FF + i) * DD + d];
}

__global__ void zcompute(const float* __restrict__ ml, const float* __restrict__ eps_i,
                         float* __restrict__ zbuf, float* __restrict__ z_o,
                         float* __restrict__ m_o, float* __restrict__ lv_o, int i)
{
    const int idx = blockIdx.x * blockDim.x + threadIdx.x;
    if (idx >= BB * LL) return;
    const int b = idx / LL, l = idx % LL;
    const float mean   = ml[b * 2 * LL + l];
    const float logvar = ml[b * 2 * LL + LL + l];
    const float zz = mean + __expf(0.5f * logvar) * eps_i[b * LL + l];
    zbuf[idx] = zz;
    const long off = ((long)b * FF + i) * LL + l;
    z_o[off] = zz; m_o[off] = mean; lv_o[off] = logvar;
}

__global__ void wnew_finalize(const float* __restrict__ whk, const float* __restrict__ zproj,
                              const float* __restrict__ posit, float* __restrict__ wbuf,
                              float* __restrict__ w_o, int i)
{
    const int idx = blockIdx.x * blockDim.x + threadIdx.x;
    if (idx >= BB * DD) return;
    const int b = idx / DD, d = idx % DD;
    const float w = whk[b * 2 * DD + d] + zproj[idx] + posit[i * DD + d];
    wbuf[((long)b * (FF + 1) + (i + 1)) * DD + d] = w;
    w_o[((long)b * FF + i) * DD + d] = w;
}

// ---------------------------------------------------------------------------
extern "C" void kernel_launch(void* const* d_in, const int* in_sizes, int n_in,
                              void* d_out, int out_size, void* d_ws, size_t ws_size,
                              hipStream_t stream)
{
    (void)in_sizes; (void)n_in; (void)out_size; (void)ws_size;
    // setup_inputs order; params flattened as a JAX pytree (dict keys sorted)
    const float* h_C = (const float*)d_in[0];
    const float* h_T = (const float*)d_in[1];
    const float* eps = (const float*)d_in[2];
    const float* fc1_b = (const float*)d_in[3];
    const float* fc1_w = (const float*)d_in[4];
    const float* fc2_b = (const float*)d_in[5];
    const float* fc2_w = (const float*)d_in[6];
    const float* ln1_b = (const float*)d_in[7];
    const float* ln1_g = (const float*)d_in[8];
    const float* ln2_b = (const float*)d_in[9];
    const float* ln2_g = (const float*)d_in[10];
    const float* posit = (const float*)d_in[11];
    const float* pm_dense_b = (const float*)d_in[12];
    const float* pm_dense_w = (const float*)d_in[13];
    const float* pm_wk_b = (const float*)d_in[14];
    const float* pm_wk_w = (const float*)d_in[15];
    const float* pm_wq_b = (const float*)d_in[16];
    const float* pm_wq_w = (const float*)d_in[17];
    const float* pm_wv_b = (const float*)d_in[18];
    const float* pm_wv_w = (const float*)d_in[19];
    const float* p1_dense_b = (const float*)d_in[20];
    const float* p1_dense_w = (const float*)d_in[21];
    const float* p1_wk_b = (const float*)d_in[22];
    const float* p1_wk_w = (const float*)d_in[23];
    const float* p1_wq_b = (const float*)d_in[24];
    const float* p1_wq_w = (const float*)d_in[25];
    const float* p1_wv_b = (const float*)d_in[26];
    const float* p1_wv_w = (const float*)d_in[27];
    const float* p2_dense_b = (const float*)d_in[28];
    const float* p2_dense_w = (const float*)d_in[29];
    const float* p2_wk_b = (const float*)d_in[30];
    const float* p2_wk_w = (const float*)d_in[31];
    const float* p2_wq_b = (const float*)d_in[32];
    const float* p2_wq_w = (const float*)d_in[33];
    const float* p2_wv_b = (const float*)d_in[34];
    const float* p2_wv_w = (const float*)d_in[35];
    const float* w0 = (const float*)d_in[36];

    float* out = (float*)d_out;
    float* w_o  = out;                              // B*F*D
    float* z_o  = out + BB * FF * DD;               // B*F*L
    float* m_o  = z_o + BB * FF * LL;
    float* lv_o = m_o + BB * FF * LL;

    // workspace bump allocator (floats)
    float* ws = (float*)d_ws;
    size_t off = 0;
    auto alloc = [&](size_t n) { float* p = ws + off; off += n; return p; };
    float* KT    = alloc((size_t)BB * TT * DD);
    float* VT    = alloc((size_t)BB * TT * DD);
    float* KC    = alloc((size_t)BB * TT * DD);
    float* VC    = alloc((size_t)BB * TT * DD);
    float* S     = alloc((size_t)BB * HH * FF * TT);
    float* Q32   = alloc((size_t)BB * FF * DD);
    float* O32   = alloc((size_t)BB * FF * DD);
    float* kall32= alloc((size_t)BB * FF * DD);
    float* wbuf  = alloc((size_t)BB * (FF + 1) * DD);
    float* Kbuf  = alloc((size_t)BB * (FF + 1) * DD);
    float* Vbuf  = alloc((size_t)BB * (FF + 1) * DD);
    float* q1b   = alloc(BB * DD);
    float* o1b   = alloc(BB * DD);
    float* o1d   = alloc(BB * DD);
    float* wbar  = alloc(BB * DD);
    float* q2b   = alloc(BB * DD);
    float* o2b   = alloc(BB * DD);
    float* o2d   = alloc(BB * DD);
    float* whk   = alloc(BB * 2 * DD);
    float* mlb   = alloc(BB * 2 * LL);
    float* zbuf  = alloc(BB * LL);
    float* zproj = alloc(BB * DD);

    const dim3 blk(32);
    const dim3 blkBig(128);
    const int ldW = (FF + 1) * DD;  // w_buf / Kbuf / Vbuf row stride per batch

    // ---------------- hoisted (loop-invariant) computation -----------------
    bcast_w0<<<16, 256, 0, stream>>>(w0, wbuf);
    wmma_gemm_big<0><<<dim3(DD/64, (BB*TT)/64), blkBig, 0, stream>>>(h_T, pm_wk_w, pm_wk_b, KT, DD, DD);
    wmma_gemm_big<0><<<dim3(DD/64, (BB*TT)/64), blkBig, 0, stream>>>(h_T, pm_wv_w, pm_wv_b, VT, DD, DD);
    wmma_gemm_big<1><<<dim3(DD/64, (BB*FF)/64), blkBig, 0, stream>>>(h_T, pm_wq_w, pm_wq_b, Q32, DD, DD);
    wmma_gemm_big<0><<<dim3(DD/64, (BB*TT)/64), blkBig, 0, stream>>>(h_C, p2_wk_w, p2_wk_b, KC, DD, DD);
    wmma_gemm_big<0><<<dim3(DD/64, (BB*TT)/64), blkBig, 0, stream>>>(h_C, p2_wv_w, p2_wv_b, VC, DD, DD);
    wmma_gemm<DD, DD><<<dim3(DD/64, 1), blk, 0, stream>>>(wbuf, p1_wk_w, p1_wk_b, Kbuf, ldW, ldW);
    wmma_gemm<DD, DD><<<dim3(DD/64, 1), blk, 0, stream>>>(wbuf, p1_wv_w, p1_wv_b, Vbuf, ldW, ldW);

    attn_scores<<<dim3(TT/64, FF/16, BB*HH), blk, 0, stream>>>(Q32, KT, S);
    softmax_rows<<<BB*HH*FF, blk, 0, stream>>>(S, TT);
    attn_av<<<dim3(FF/16, BB*HH), blk, 0, stream>>>(S, VT, O32);
    wmma_gemm_big<0><<<dim3(DD/64, (BB*FF)/64), blkBig, 0, stream>>>(O32, pm_dense_w, pm_dense_b, kall32, DD, DD);

    // ------------------------------ scan -----------------------------------
    for (int i = 0; i < FF; ++i) {
        const float* wrow = wbuf + i * DD;                          // (B,1,D), ld=ldW
        wmma_gemm<DD, DD><<<dim3(DD/64, 1), blk, 0, stream>>>(wrow, p1_wq_w, p1_wq_b, q1b, ldW, DD);
        attn_small<<<BB*HH, blk, 0, stream>>>(q1b, Kbuf, Vbuf, o1b, i + 1);
        wmma_gemm<DD, DD><<<dim3(DD/64, 1), blk, 0, stream>>>(o1b, p1_dense_w, p1_dense_b, o1d, DD, DD);
        ln_resid<<<BB, blk, 0, stream>>>(wrow, ldW, o1d, ln1_g, ln1_b, wbar, DD);

        wmma_gemm<DD, DD><<<dim3(DD/64, 1), blk, 0, stream>>>(wbar, p2_wq_w, p2_wq_b, q2b, DD, DD);
        attn_long<<<BB*HH, blk, 0, stream>>>(q2b, KC, VC, o2b);
        wmma_gemm<DD, DD><<<dim3(DD/64, 1), blk, 0, stream>>>(o2b, p2_dense_w, p2_dense_b, o2d, DD, DD);
        ln_resid<<<BB, blk, 0, stream>>>(wbar, DD, o2d, ln2_g, ln2_b, whk, 2*DD);

        copy_ki<<<16, 256, 0, stream>>>(kall32, whk, i);
        wmma_gemm<2*DD, 2*LL><<<dim3((2*LL)/64, 1), blk, 0, stream>>>(whk, fc1_w, fc1_b, mlb, 2*DD, 2*LL);
        zcompute<<<4, 256, 0, stream>>>(mlb, eps + (long)i * BB * LL, zbuf, z_o, m_o, lv_o, i);
        wmma_gemm<LL, DD><<<dim3(DD/64, 1), blk, 0, stream>>>(zbuf, fc2_w, fc2_b, zproj, LL, DD);
        wnew_finalize<<<16, 256, 0, stream>>>(whk, zproj, posit, wbuf, w_o, i);

        // incremental K/V projection of the newly appended w_buf row
        const float* wnew = wbuf + (i + 1) * DD;
        wmma_gemm<DD, DD><<<dim3(DD/64, 1), blk, 0, stream>>>(wnew, p1_wk_w, p1_wk_b, Kbuf + (i + 1) * DD, ldW, ldW);
        wmma_gemm<DD, DD><<<dim3(DD/64, 1), blk, 0, stream>>>(wnew, p1_wv_w, p1_wv_b, Vbuf + (i + 1) * DD, ldW, ldW);
    }
}
